// StringGNNTailExtended_6923487282041
// MI455X (gfx1250) — matrix operand order repaired and checked by reading, not compiled, not run on hardware
//
#include <hip/hip_runtime.h>
#include <hip/hip_bf16.h>
#include <stdint.h>

// ---------------- types ----------------
typedef __attribute__((ext_vector_type(16))) __bf16          v16bf;
typedef __attribute__((ext_vector_type(16))) unsigned short  v16u;
typedef __attribute__((ext_vector_type(8)))  float           v8f;
typedef __attribute__((ext_vector_type(4)))  unsigned int    v4u;
typedef __attribute__((ext_vector_type(8)))  int             v8i;
typedef __attribute__((ext_vector_type(4)))  int             v4i;

#define D_DIM 256

// fp32 -> bf16 (round to nearest even), raw bits
__device__ __forceinline__ unsigned short f2bf(float f) {
    unsigned int u = __float_as_uint(f);
    unsigned int r = u + 0x7fffu + ((u >> 16) & 1u);
    return (unsigned short)(r >> 16);
}

// ---------------- utility kernels ----------------
__global__ void k_zero(float* __restrict__ p, long long n) {
    long long i = (long long)blockIdx.x * blockDim.x + threadIdx.x;
    if (i < n) p[i] = 0.0f;
}

__global__ void k_deg(const long long* __restrict__ ei, const float* __restrict__ w,
                      float* __restrict__ deg_out, float* __restrict__ deg_in,
                      long long nE) {
    long long e = (long long)blockIdx.x * blockDim.x + threadIdx.x;
    if (e >= nE) return;
    atomicAdd(&deg_out[ei[e]],      w[e]);
    atomicAdd(&deg_in [ei[nE + e]], w[e]);
}

__global__ void k_norm(const long long* __restrict__ ei, const float* __restrict__ w,
                       const float* __restrict__ deg_out, const float* __restrict__ deg_in,
                       float* __restrict__ norm, long long nE) {
    long long e = (long long)blockIdx.x * blockDim.x + threadIdx.x;
    if (e >= nE) return;
    norm[e] = w[e] * rsqrtf(deg_out[ei[e]] * deg_in[ei[nE + e]] + 1e-12f);
}

// fp32 features -> bf16 features (GEMM A operand lives in bf16 to halve HBM traffic)
__global__ void k_x2bf(const float* __restrict__ x, unsigned short* __restrict__ xb, long long n) {
    long long i = (long long)blockIdx.x * blockDim.x + threadIdx.x;
    if (i < n) xb[i] = f2bf(x[i]);
}

// Repack a 256x256 fp32 weight into bf16 *fragment order*:
// Wpk[(((c*8)+kk)*32 + lane)*16 + i] = bf16( W[(kk*32 + (lane>>4)*16 + i)*256 + c*16 + (lane&15)] )
__global__ void k_pack_w(const float* __restrict__ W, unsigned short* __restrict__ Wpk) {
    int t = blockIdx.x * blockDim.x + threadIdx.x;     // (c, kk, lane)
    if (t >= 16 * 8 * 32) return;
    int lane = t & 31;
    int kk   = (t >> 5) & 7;
    int c    = t >> 8;
    int kb   = (kk << 5) + ((lane >> 4) << 4);
    int col  = (c << 4) + (lane & 15);
    v16u o;
#pragma unroll
    for (int i = 0; i < 16; ++i)
        o[i] = f2bf(W[(long long)(kb + i) * D_DIM + col]);
    *(v16u*)(Wpk + ((long long)t << 4)) = o;
}

// Per-edge gather + scale + atomic scatter-add.  One thread per (edge, 4-float chunk).
__global__ void k_scatter(const float* __restrict__ xt, const long long* __restrict__ ei,
                          const float* __restrict__ norm, float* __restrict__ agg,
                          long long nE) {
    long long t = (long long)blockIdx.x * blockDim.x + threadIdx.x;
    long long total = nE << 6;                 // E * (256/4)
    if (t >= total) return;
    long long e = t >> 6;
    int c = (int)(t & 63) << 2;
    long long s = ei[e];
    long long d = ei[nE + e];
    float nw = norm[e];
    const float4 v = *(const float4*)(xt + s * D_DIM + c);
    float* base = agg + d * D_DIM + c;
    atomicAdd(base + 0, v.x * nw);
    atomicAdd(base + 1, v.y * nw);
    atomicAdd(base + 2, v.z * nw);
    atomicAdd(base + 3, v.w * nw);
}

// h = relu(agg + b), written as bf16 (next layer's GEMM A operand)
__global__ void k_bias_relu_bf16(const float* __restrict__ agg, const float* __restrict__ b,
                                 unsigned short* __restrict__ hb, long long n) {
    long long i = (long long)blockIdx.x * blockDim.x + threadIdx.x;
    if (i >= n) return;
    hb[i] = f2bf(fmaxf(agg[i] + b[i & (D_DIM - 1)], 0.0f));
}

// ---------------- WMMA GEMM: Y[nrows x 256] = Xb[nrows x 256](bf16) @ W[256 x 256] (+bias)
// Block = 512 threads = 16 waves.  Block owns 32 rows; wave w owns columns [16w, 16w+16).
// A tile is DMA'd into LDS by the Tensor Data Mover with hardware row padding
// (512B rows + 16B pad = 528B pitch) and hardware zero-fill past nrows.
#define AP 264   // LDS row pitch in ushorts (528 B) -- must match the TDM pad config below

__global__ void __launch_bounds__(512)
k_gemm_wmma(const unsigned short* __restrict__ Xb, const unsigned short* __restrict__ Wpk,
            const float* __restrict__ bias, float* __restrict__ Y, int nrows) {
    __shared__ unsigned short As[32][AP];

    const int tid  = threadIdx.x;
    const int row0 = blockIdx.x << 5;

    // ---- TDM: one tensor_load_to_lds per block (wave 0 issues; EXEC-independent op) ----
    if (tid < 32) {
        unsigned long long ga = (unsigned long long)(const void*)(Xb + (size_t)row0 * D_DIM);
        unsigned lds = (unsigned)(uintptr_t)(&As[0][0]);
        unsigned rem = (unsigned)(nrows - row0);       // tensor extent from tile start (HW zero-fills OOB)

        // D# group 0: count=1, lds_addr, global_addr[56:0], type=2
        v4u g0;
        g0[0] = 1u;
        g0[1] = lds;
        g0[2] = (unsigned)(ga & 0xffffffffu);
        g0[3] = (unsigned)((ga >> 32) & 0x01ffffffu) | (2u << 30);

        // D# group 1: data_size=8B; pad_enable, pad_interval=128 DW (512B), pad_amount=4 DW (16B);
        // tensor_dim0=64 (8B units/row), tensor_dim1=rem rows, tile 64x32, dim0 stride=64 (contig rows)
        v8i g1;
        g1[0] = (3 << 16) | (1 << 20) | (6 << 22) | (3 << 25);
        g1[1] = (int)(64u << 16);                                   // tensor_dim0 lo16
        g1[2] = (int)((rem & 0xffffu) << 16);                       // tensor_dim0 hi16 | tensor_dim1 lo16
        g1[3] = (int)(((rem >> 16) & 0xffffu) | (64u << 16));       // tensor_dim1 hi16 | tile_dim0=64
        g1[4] = 32;                                                 // tile_dim1=32, tile_dim2=0
        g1[5] = 64;                                                 // tensor_dim0_stride lo32
        g1[6] = 0;
        g1[7] = 0;

        v4i g2 = {0, 0, 0, 0};
        v4i g3 = {0, 0, 0, 0};
        v8i g4 = {0, 0, 0, 0, 0, 0, 0, 0};
        __builtin_amdgcn_tensor_load_to_lds(g0, g1, g2, g3, g4, 0);
        __builtin_amdgcn_s_wait_tensorcnt(0);
    }
    __syncthreads();

    const int wave = tid >> 5;       // 0..15 -> 16-column tile
    const int lane = tid & 31;
    const int half = lane >> 4;
    const int mn   = lane & 15;
    const int col  = (wave << 4) + mn;

    v8f acc0 = {};
    v8f acc1 = {};

#pragma unroll
    for (int kk = 0; kk < 8; ++kk) {
        const int kbase = kk << 5;

        // A fragments: element i -> K = half*16 + i (matches contiguous B packing)
        v16u au0, au1;
#pragma unroll
        for (int i = 0; i < 16; ++i) {
            int ka = kbase + (half << 4) + i;
            au0[i] = As[mn][ka];           // rows 0..15
            au1[i] = As[16 + mn][ka];      // rows 16..31
        }

        // B fragment: contiguous 32B per lane, lane-consecutive (pre-swizzled in workspace)
        const v16u bu = *(const v16u*)(Wpk + ((long long)(((wave << 3) + kk) << 5 | lane) << 4));

        v16bf a0 = __builtin_bit_cast(v16bf, au0);
        v16bf a1 = __builtin_bit_cast(v16bf, au1);
        v16bf b  = __builtin_bit_cast(v16bf, bu);
        acc0 = __builtin_amdgcn_wmma_f32_16x16x32_bf16(false, a0, false, b, (short)0, acc0, false, false);
        acc1 = __builtin_amdgcn_wmma_f32_16x16x32_bf16(false, a1, false, b, (short)0, acc1, false, false);
    }

    // D layout: VGPR r -> M = half*8 + r, N = col
    const float bv = (bias != nullptr) ? bias[col] : 0.0f;
    const int m0 = half << 3;
#pragma unroll
    for (int r = 0; r < 8; ++r) {
        int rowA = row0 + m0 + r;
        int rowB = rowA + 16;
        if (rowA < nrows) Y[(long long)rowA * D_DIM + col] = acc0[r] + bv;
        if (rowB < nrows) Y[(long long)rowB * D_DIM + col] = acc1[r] + bv;
    }
}

// ---------------- host-side orchestration ----------------
static inline unsigned blocks_for(long long n, int bs) {
    return (unsigned)((n + bs - 1) / bs);
}

extern "C" void kernel_launch(void* const* d_in, const int* in_sizes, int n_in,
                              void* d_out, int out_size, void* d_ws, size_t ws_size,
                              hipStream_t stream) {
    const float*     h5 = (const float*)d_in[0];
    const long long* ei = (const long long*)d_in[1];   // int64 [2, E]
    const float*     ew = (const float*)d_in[2];
    const float*     W6 = (const float*)d_in[3];
    const float*     b6 = (const float*)d_in[4];
    const float*     W7 = (const float*)d_in[5];
    const float*     b7 = (const float*)d_in[6];
    const float*     Wp = (const float*)d_in[7];
    const float*     bp = (const float*)d_in[8];

    const long long N  = (long long)in_sizes[0] / D_DIM;
    const long long E  = (long long)in_sizes[2];
    const long long ND = N * D_DIM;
    const long long WW = (long long)D_DIM * D_DIM;

    // workspace carve-up (256B aligned)
    char* ws = (char*)d_ws;
    size_t off = 0;
    auto carve = [&](size_t bytes) -> char* {
        char* p = ws + off;
        off += (bytes + 255) & ~(size_t)255;
        return p;
    };
    float*          deg_out = (float*)carve(N * sizeof(float));
    float*          deg_in  = (float*)carve(N * sizeof(float));
    float*          norm    = (float*)carve(E * sizeof(float));
    unsigned short* Wb6     = (unsigned short*)carve(WW * sizeof(unsigned short));
    unsigned short* Wb7     = (unsigned short*)carve(WW * sizeof(unsigned short));
    unsigned short* Wbp     = (unsigned short*)carve(WW * sizeof(unsigned short));
    unsigned short* Xb      = (unsigned short*)carve(ND * sizeof(unsigned short)); // bf16 features
    float*          B1      = (float*)carve(ND * sizeof(float));   // xt (fp32 for atomics)
    float*          B2      = (float*)carve(ND * sizeof(float));   // agg
    float*          out     = (float*)d_out;

    const unsigned gemm_grid = blocks_for(N, 32);

    // --- normalization ---
    k_zero<<<blocks_for(2 * N, 256), 256, 0, stream>>>(deg_out, 2 * N); // deg_out+deg_in contiguous
    k_deg<<<blocks_for(E, 256), 256, 0, stream>>>(ei, ew, deg_out, deg_in, E);
    k_norm<<<blocks_for(E, 256), 256, 0, stream>>>(ei, ew, deg_out, deg_in, norm, E);

    // --- weight repack + feature conversion (bf16) ---
    k_pack_w<<<blocks_for(16 * 8 * 32, 256), 256, 0, stream>>>(W6, Wb6);
    k_pack_w<<<blocks_for(16 * 8 * 32, 256), 256, 0, stream>>>(W7, Wb7);
    k_pack_w<<<blocks_for(16 * 8 * 32, 256), 256, 0, stream>>>(Wp, Wbp);
    k_x2bf<<<blocks_for(ND, 256), 256, 0, stream>>>(h5, Xb, ND);

    // --- layer 1: xt = h5 @ W6 ; agg = scatter ; h6 = relu(agg + b6) -> Xb ---
    k_gemm_wmma<<<gemm_grid, 512, 0, stream>>>(Xb, Wb6, nullptr, B1, (int)N);
    k_zero<<<blocks_for(ND, 256), 256, 0, stream>>>(B2, ND);
    k_scatter<<<blocks_for(E << 6, 256), 256, 0, stream>>>(B1, ei, norm, B2, E);
    k_bias_relu_bf16<<<blocks_for(ND, 256), 256, 0, stream>>>(B2, b6, Xb, ND);

    // --- layer 2: xt = h6 @ W7 ; agg ; h7 = relu(agg + b7) -> Xb ---
    k_gemm_wmma<<<gemm_grid, 512, 0, stream>>>(Xb, Wb7, nullptr, B1, (int)N);
    k_zero<<<blocks_for(ND, 256), 256, 0, stream>>>(B2, ND);
    k_scatter<<<blocks_for(E << 6, 256), 256, 0, stream>>>(B1, ei, norm, B2, E);
    k_bias_relu_bf16<<<blocks_for(ND, 256), 256, 0, stream>>>(B2, b7, Xb, ND);

    // --- head: out = h7 @ Wp + bp ---
    k_gemm_wmma<<<gemm_grid, 512, 0, stream>>>(Xb, Wbp, bp, out, (int)N);
}
